// MultiHeadAttention_73693048865167
// MI455X (gfx1250) — matrix operand order repaired and checked by reading, not compiled, not run on hardware
//
#include <hip/hip_runtime.h>
#include <hip/hip_bf16.h>

typedef __attribute__((ext_vector_type(16))) _Float16 v16h;
typedef __attribute__((ext_vector_type(8)))  _Float16 v8h;
typedef __attribute__((ext_vector_type(8)))  float    v8f;

#define D_MODEL 1024
#define NHEADS  16
#define DEPTH   64
#define BATCH   2
#define SEQ     2048
#define MTOT    (BATCH * SEQ)      // 4096 rows total
#define AW      8                  // waves per block (256 threads, wave32)

union U16h { v16h v; v8h h[2]; _Float16 e[16]; };
union U8f  { v8f v; float e[8]; };

static __device__ __forceinline__ v8f wmma_f16(v16h a, v16h b, v8f c) {
  // D = A(16x32 f16) x B(32x16 f16) + C(16x16 f32)
  return __builtin_amdgcn_wmma_f32_16x16x32_f16(
      /*neg_a=*/false, a, /*neg_b=*/false, b,
      /*c_mod=*/(short)0, c, /*reuse_a=*/false, /*reuse_b=*/false);
}

// gfx1250 async global->LDS copy (ASYNCcnt-tracked). Inline asm: portable
// across ROCm7.2 / amdgpu-toolchain builtin arity differences.
static __device__ __forceinline__ void async_load_b128(unsigned lds_off,
                                                       const void* gptr) {
  unsigned long long ga = (unsigned long long)gptr;
  asm volatile("global_load_async_to_lds_b128 %0, %1, off"
               :: "v"(lds_off), "v"(ga) : "memory");
}
static __device__ __forceinline__ void wait_async_le2() {
  asm volatile("s_wait_asynccnt 0x2" ::: "memory");
}
static __device__ __forceinline__ void wait_async_le0() {
  asm volatile("s_wait_asynccnt 0x0" ::: "memory");
}

// ---------------------------------------------------------------------------
// Kernel 1: W [K=1024][N=1024] f32  ->  Wt [N][K] f16  (transpose + convert)
// ---------------------------------------------------------------------------
__global__ __launch_bounds__(256) void transpose_w_kernel(
    const float* __restrict__ W, _Float16* __restrict__ Wt) {
  int idx = blockIdx.x * 256 + threadIdx.x;   // 1M elements
  int k = idx >> 10;
  int n = idx & 1023;
  Wt[n * D_MODEL + k] = (_Float16)W[idx];
}

// ---------------------------------------------------------------------------
// Kernel 2: WMMA GEMM   C[M x 1024] = A[M x 1024] @ W + bias
//   AMODE: 0 = A is f32 (converted on load), 1 = A is f16
//   SMODE: 0 = f32 plain [M][1024]
//          1 = f16 head-split  dst[((b*H+h)*S + s)*64 + d]   (Q, K)
//          2 = f16 V-transpose dst[((b*H+h)*64 + d)*S + s]   (V)
// One wave computes a 16(M) x 64(N) tile; K-steps of 32; 4 WMMA accumulators.
// Block's 8 waves share the same N-tile (B fragments amortized via WGP$/L2)
// and cover 8 consecutive M-tiles.
// ---------------------------------------------------------------------------
template <int AMODE, int SMODE>
__global__ __launch_bounds__(256) void gemm_kernel(
    const void* __restrict__ Av, const _Float16* __restrict__ Wt,
    const float* __restrict__ bias, void* __restrict__ Out) {
  const int lane = threadIdx.x & 31;
  const int wid  = blockIdx.x * AW + (threadIdx.x >> 5);   // 4096 waves
  const int mt   = wid & 255;        // 0..255  (consecutive within a block)
  const int nt   = wid >> 8;         // 0..15   (uniform within a block)
  const int m0   = mt * 16;
  const int n0w  = nt * 64;
  const int half = lane >> 4;
  const int ln   = lane & 15;

  v8f acc[4] = {{}, {}, {}, {}};

  const int arow = m0 + ln;
  for (int kk = 0; kk < D_MODEL; kk += 32) {
    // ---- A fragment (16x32 f16): lane<16 holds K 0-15, lane>=16 K 16-31
    U16h a;
    const int kbase = kk + half * 16;
    if (AMODE == 0) {
      const float* A  = (const float*)Av;
      const float* p0 = A + arow * D_MODEL + kbase;
#pragma unroll
      for (int i = 0; i < 8; ++i) {
        a.e[i]     = (_Float16)p0[i];
        a.e[8 + i] = (_Float16)p0[8 + i];
      }
    } else {
      const _Float16* A  = (const _Float16*)Av;
      const _Float16* p0 = A + arow * D_MODEL + kbase;
      a.h[0] = *(const v8h*)p0;
      a.h[1] = *(const v8h*)(p0 + 8);
    }
    // ---- B fragments (32x16 f16): lane = N col, halves split K 0-15 / 16-31
#pragma unroll
    for (int j = 0; j < 4; ++j) {
      const int n   = n0w + j * 16 + ln;
      const v16h bf = *(const v16h*)(Wt + n * D_MODEL + kk + half * 16);
      acc[j] = wmma_f16(a.v, bf, acc[j]);
    }
  }

  // ---- epilogue: bias + store (C layout: row M = half*8+r, col N = ln)
#pragma unroll
  for (int j = 0; j < 4; ++j) {
    const int n    = n0w + j * 16 + ln;
    const float bb = bias[n];
    U8f c; c.v = acc[j];
#pragma unroll
    for (int r = 0; r < 8; ++r) {
      const int mrow  = m0 + half * 8 + r;
      const float val = c.e[r] + bb;
      if (SMODE == 0) {
        ((float*)Out)[mrow * D_MODEL + n] = val;
      } else {
        const int b  = mrow >> 11;       // / SEQ
        const int s  = mrow & (SEQ - 1);
        const int hh = n >> 6;
        const int dd = n & 63;
        if (SMODE == 1)
          ((_Float16*)Out)[(((b * NHEADS + hh) * SEQ) + s) * DEPTH + dd] = (_Float16)val;
        else
          ((_Float16*)Out)[(((b * NHEADS + hh) * DEPTH) + dd) * SEQ + s] = (_Float16)val;
      }
    }
  }
}

// ---------------------------------------------------------------------------
// Kernel 3: causal flash attention, block-cooperative.
//   block = 8 waves = one (b,h) x 8 consecutive 16-query tiles.
//   K/V 32-key tiles staged in LDS via double-buffered async b128 copies.
//   Qp, Kp: f16 [b,h,s,d]; Vt: f16 [b,h,d,s]; Outp: f16 [b, s, h*64+d]
// ---------------------------------------------------------------------------
__global__ __launch_bounds__(256) void attn_kernel(
    const _Float16* __restrict__ Qp, const _Float16* __restrict__ Kp,
    const _Float16* __restrict__ Vt, _Float16* __restrict__ Outp) {
  __shared__ __align__(128) _Float16 ldsK[2][32 * 64];  // [buf][key][d]  4 KB each
  __shared__ __align__(128) _Float16 ldsV[2][64 * 32];  // [buf][d][key]  4 KB each
  __shared__ __align__(128) _Float16 ldsP[AW][16 * 40]; // per-wave P slab

  const int tid   = threadIdx.x;
  const int lane  = tid & 31;
  const int wslot = tid >> 5;
  const int blk   = blockIdx.x;        // 512 blocks
  const int qg    = blk & 15;          // 128-query group
  const int h     = (blk >> 4) & 15;
  const int b     = blk >> 8;
  const int q0    = (qg * AW + wslot) * 16;
  const int half  = lane >> 4;
  const int ln    = lane & 15;
  _Float16* P = &ldsP[wslot][0];

  const _Float16* Qb = Qp + (size_t)((b * NHEADS + h) * SEQ) * DEPTH;
  const char* KbB = (const char*)(Kp + (size_t)((b * NHEADS + h) * SEQ) * DEPTH);
  const char* VbB = (const char*)(Vt + (size_t)((b * NHEADS + h) * DEPTH) * SEQ);

  // LDS byte offsets (flat->LDS: low 32 bits of the generic pointer)
  const unsigned offK = (unsigned)(size_t)&ldsK[0][0];
  const unsigned offV = (unsigned)(size_t)&ldsV[0][0];
  const int vrow  = tid >> 2;          // d row for V tile copy
  const int vpart = tid & 3;           // 16B chunk within 64B row

  // Q fragments for d-chunks {0..31}, {32..63}; resident whole kernel
  U16h qa[2];
  {
    const _Float16* qrow = Qb + (q0 + ln) * DEPTH;
#pragma unroll
    for (int t = 0; t < 2; ++t) {
      const int kb = t * 32 + half * 16;
      qa[t].h[0] = *(const v8h*)(qrow + kb);
      qa[t].h[1] = *(const v8h*)(qrow + kb + 8);
    }
  }

  v8f o[4] = {{}, {}, {}, {}};   // 16 q-rows x 64 d accumulators
  U8f mrow, lrow;
#pragma unroll
  for (int r = 0; r < 8; ++r) { mrow.e[r] = -3.0e38f; lrow.e[r] = 0.0f; }

  const int kend_wave  = q0 + 16;              // causal bound for this wave
  const int kend_block = qg * 128 + 128;       // causal bound for last wave

  // issue one 32-key K tile (4KB contiguous) + V tile (64 x 64B rows)
  auto issue = [&](int kb) {
    const unsigned bi = ((unsigned)kb >> 5) & 1u;
    async_load_b128(offK + bi * 4096u + tid * 16,
                    KbB + (size_t)kb * 128 + tid * 16);
    async_load_b128(offV + bi * 4096u + vrow * 64 + vpart * 16,
                    VbB + (size_t)vrow * (SEQ * 2) + (size_t)kb * 2 + vpart * 16);
  };

  issue(0);
  for (int kb = 0; kb < kend_block; kb += 32) {
    if (kb + 32 < kend_block) { issue(kb + 32); wait_async_le2(); }
    else                      { wait_async_le0(); }
    __syncthreads();                           // tile resident for whole block

    // wave-uniform causal skip; force a scalar branch (keeps EXEC untouched)
    if (__builtin_amdgcn_readfirstlane((int)(kb < kend_wave))) {
      const int bi = (kb >> 5) & 1;
      const _Float16* tK = &ldsK[bi][0];
      const _Float16* tV = &ldsV[bi][0];

      // ---- logits: two 16x16 C tiles covering 32 keys, K-reduced over d=64
      v8f c[2] = {{}, {}};
#pragma unroll
      for (int n0 = 0; n0 < 2; ++n0) {
#pragma unroll
        for (int t = 0; t < 2; ++t) {
          const v16h bk = *(const v16h*)(tK + (n0 * 16 + ln) * 64 + t * 32 + half * 16);
          c[n0] = wmma_f16(qa[t].v, bk, c[n0]);
        }
      }
      // ---- scale + causal mask + online softmax (rows across 16-lane halves)
      U8f c0, c1; c0.v = c[0]; c1.v = c[1];
      const int qbase = q0 + half * 8;
#pragma unroll
      for (int r = 0; r < 8; ++r) {
        const int qg2 = qbase + r;
        float x0 = c0.e[r] * 0.125f + ((kb + ln      > qg2) ? -1.0e9f : 0.0f);
        float x1 = c1.e[r] * 0.125f + ((kb + 16 + ln > qg2) ? -1.0e9f : 0.0f);
        float tm = fmaxf(x0, x1);
        tm = fmaxf(tm, __shfl_xor(tm, 1));
        tm = fmaxf(tm, __shfl_xor(tm, 2));
        tm = fmaxf(tm, __shfl_xor(tm, 4));
        tm = fmaxf(tm, __shfl_xor(tm, 8));
        const float mnew  = fmaxf(mrow.e[r], tm);
        const float alpha = __expf(mrow.e[r] - mnew);
        const float p0    = __expf(x0 - mnew);
        const float p1    = __expf(x1 - mnew);
        float ps = p0 + p1;
        ps += __shfl_xor(ps, 1);
        ps += __shfl_xor(ps, 2);
        ps += __shfl_xor(ps, 4);
        ps += __shfl_xor(ps, 8);
        lrow.e[r] = lrow.e[r] * alpha + ps;
        mrow.e[r] = mnew;
#pragma unroll
        for (int t = 0; t < 4; ++t) o[t][r] *= alpha;
        const int prow = half * 8 + r;         // stash P in C layout
        P[prow * 40 + ln]      = (_Float16)p0;
        P[prow * 40 + 16 + ln] = (_Float16)p1;
      }
      asm volatile("s_wait_dscnt 0" ::: "memory");   // same-wave DS drain
      // ---- read P back as an A fragment (16x32 f16)
      U16h pa;
      {
        const _Float16* pr = P + ln * 40 + half * 16;
        pa.h[0] = *(const v8h*)pr;
        pa.h[1] = *(const v8h*)(pr + 8);
      }
      // ---- O += P @ V : 4 WMMAs cover d = 0..63, K = 32 keys
#pragma unroll
      for (int t = 0; t < 4; ++t) {
        const v16h bv = *(const v16h*)(tV + (t * 16 + ln) * 32 + half * 16);
        o[t] = wmma_f16(pa.v, bv, o[t]);
      }
    }
    __syncthreads();   // all waves done reading before buffer reuse
  }

  // ---- normalize + store f16 [b, s, h*64 + d]
  U8f inv;
#pragma unroll
  for (int r = 0; r < 8; ++r) inv.e[r] = 1.0f / lrow.e[r];
#pragma unroll
  for (int t = 0; t < 4; ++t) {
    U8f ot; ot.v = o[t];
#pragma unroll
    for (int r = 0; r < 8; ++r) {
      const int s   = q0 + half * 8 + r;
      const int col = h * DEPTH + t * 16 + ln;
      Outp[(size_t)(b * SEQ + s) * D_MODEL + col] = (_Float16)(ot.e[r] * inv.e[r]);
    }
  }
}

// ---------------------------------------------------------------------------
extern "C" void kernel_launch(void* const* d_in, const int* in_sizes, int n_in,
                              void* d_out, int out_size, void* d_ws, size_t ws_size,
                              hipStream_t stream) {
  const float* v_in = (const float*)d_in[0];
  const float* k_in = (const float*)d_in[1];
  const float* q_in = (const float*)d_in[2];
  // d_in[3] = mask: causal, computed analytically in-kernel
  const float* Wq = (const float*)d_in[4];
  const float* bq = (const float*)d_in[5];
  const float* Wk = (const float*)d_in[6];
  const float* bk = (const float*)d_in[7];
  const float* Wv = (const float*)d_in[8];
  const float* bv = (const float*)d_in[9];
  const float* Wo = (const float*)d_in[10];
  const float* bo = (const float*)d_in[11];

  char* ws = (char*)d_ws;
  size_t off = 0;
  auto alloc = [&](size_t bytes) -> void* {
    void* p = ws + off;
    off = (off + bytes + 255) & ~(size_t)255;
    return p;
  };
  const size_t wbytes = (size_t)D_MODEL * D_MODEL * sizeof(_Float16);  // 2 MB
  const size_t hbytes = (size_t)MTOT * D_MODEL * sizeof(_Float16);     // 8 MB
  _Float16* WqT = (_Float16*)alloc(wbytes);
  _Float16* WkT = (_Float16*)alloc(wbytes);
  _Float16* WvT = (_Float16*)alloc(wbytes);
  _Float16* WoT = (_Float16*)alloc(wbytes);
  _Float16* Qp  = (_Float16*)alloc(hbytes);   // [b,h,s,d]
  _Float16* Kp  = (_Float16*)alloc(hbytes);   // [b,h,s,d]
  _Float16* Vtr = (_Float16*)alloc(hbytes);   // [b,h,d,s]
  _Float16* AO  = (_Float16*)alloc(hbytes);   // [b,s,h*64+d]

  // 1) weights -> f16, transposed
  transpose_w_kernel<<<4096, 256, 0, stream>>>(Wq, WqT);
  transpose_w_kernel<<<4096, 256, 0, stream>>>(Wk, WkT);
  transpose_w_kernel<<<4096, 256, 0, stream>>>(Wv, WvT);
  transpose_w_kernel<<<4096, 256, 0, stream>>>(Wo, WoT);

  // 2) QKV projections (WMMA), head-split outputs
  gemm_kernel<0, 1><<<512, 256, 0, stream>>>(q_in, WqT, bq, Qp);
  gemm_kernel<0, 1><<<512, 256, 0, stream>>>(k_in, WkT, bk, Kp);
  gemm_kernel<0, 2><<<512, 256, 0, stream>>>(v_in, WvT, bv, Vtr);

  // 3) causal flash attention (WMMA + async-to-LDS double buffering)
  attn_kernel<<<512, 256, 0, stream>>>(Qp, Kp, Vtr, AO);

  // 4) output projection (WMMA), f32 result
  gemm_kernel<1, 0><<<512, 256, 0, stream>>>(AO, WoT, bo, (float*)d_out);
}
